// MultiScaleDeformConv_31267361915086
// MI455X (gfx1250) — matrix operand order repaired and checked by reading, not compiled, not run on hardware
//
#include <hip/hip_runtime.h>
#include <hip/hip_bf16.h>

#define DEVINL __device__ __forceinline__

typedef __attribute__((ext_vector_type(16))) __bf16 v16bf;
typedef __attribute__((ext_vector_type(8)))  __bf16 v8bf;
typedef __attribute__((ext_vector_type(8)))  float  v8f;
typedef unsigned int ui32x4 __attribute__((ext_vector_type(4)));
typedef int          i32x4  __attribute__((ext_vector_type(4)));
typedef int          i32x8  __attribute__((ext_vector_type(8)));

#if defined(__AMDGCN__) && __has_builtin(__builtin_amdgcn_tensor_load_to_lds)
#define HAVE_TDM 1
#else
#define HAVE_TDM 0
#endif

// ---------------------------------------------------------------------------
// WMMA helpers (CDNA5 / gfx1250, wave32).
// v_wmma_f32_16x16x32_bf16: D(16x16 f32) = A(16x32 bf16) * B(32x16 bf16) + C
// Layouts used here (contiguous k order, shared by A and B):
//   A: lane L holds row m=L&15; half h -> k = kk + h + 16*(L>>4)
//   B: lane L holds col n=L&15; half h -> k = kk + h + 16*(L>>4)
//   C/D: lane L col n=L&15; vgpr r -> row m = r + 8*(L>>4)
// ---------------------------------------------------------------------------
DEVINL v8f wmma_bf16(v16bf a, v16bf b, v8f c) {
  return __builtin_amdgcn_wmma_f32_16x16x32_bf16(
      /*neg_a=*/false, a, /*neg_b=*/false, b,
      /*c_mod=*/(short)0, c, /*reuse_a=*/false, /*reuse_b=*/false);
}

DEVINL v8f vzero8() {
  v8f z;
#pragma unroll
  for (int i = 0; i < 8; ++i) z[i] = 0.0f;
  return z;
}

// Load one A fragment from a row-major (M x Kd) bf16 matrix (global or LDS
// via generic pointer). Kd % 32 == 0 so the two 8-half runs are 16B aligned.
DEVINL v16bf load_a_rowmajor(const __bf16* __restrict__ W, int row, int Kd,
                             int kk, int g, bool valid) {
  v16bf a;
  if (valid) {
    const __bf16* base = W + (size_t)row * Kd + kk + (g ? 16 : 0);
    v8bf lo = *(const v8bf*)(base);
    v8bf hi = *(const v8bf*)(base + 8);
#pragma unroll
    for (int i = 0; i < 8; ++i) { a[i] = lo[i]; a[8 + i] = hi[i]; }
  } else {
#pragma unroll
    for (int i = 0; i < 16; ++i) a[i] = (__bf16)0.0f;
  }
  return a;
}

// ---------------------------------------------------------------------------
// Tensor Data Mover: stage a 2D tile (tile_w x tile_h elements, row stride
// `row_stride` elements) from global memory into LDS. D# layout per
// CDNA5 ISA. This toolchain exposes the 6-operand
// builtin: (g0 v4u, g1 v8i, g2 v4i, g3 v4i, g4 v8i, cpol). 2D tiles only
// need groups 0/1; trailing descriptor groups are zero. Issued by wave 0;
// completion via TENSORcnt (s_wait_tensorcnt) then workgroup barrier.
// ---------------------------------------------------------------------------
#if HAVE_TDM
DEVINL unsigned lds_addr_of(const void* p) {
  return (unsigned)(unsigned long long)
      (__attribute__((address_space(3))) const void*)p;
}

DEVINL void tdm_load_2d(const void* gsrc, unsigned lds_byte_addr,
                        unsigned tile_w, unsigned tile_h,
                        unsigned long long row_stride_elems,
                        unsigned data_size_log2) {
  unsigned long long ga = (unsigned long long)(size_t)gsrc;
  unsigned long long d0s = row_stride_elems;                 // 48-bit
  unsigned long long d1s = row_stride_elems * tile_h;        // 48-bit
  ui32x4 g0;
  g0[0] = 1u;                                   // count=1, user descriptor
  g0[1] = lds_byte_addr;                        // lds_addr [63:32]
  g0[2] = (unsigned)ga;                         // global_addr lo
  g0[3] = (unsigned)((ga >> 32) & 0x01FFFFFFu)  // global_addr hi (57-bit)
          | (2u << 30);                         // type=2 ("image")
  i32x8 g1;
  g1[0] = (int)(data_size_log2 << 16);          // wg_mask=0, data_size
  g1[1] = (int)((tile_w & 0xFFFFu) << 16);      // tensor_dim0[15:0]
  g1[2] = (int)((tile_w >> 16) | ((tile_h & 0xFFFFu) << 16));  // d0 hi, d1 lo
  g1[3] = (int)((tile_h >> 16) | (tile_w << 16));              // d1 hi, tile_dim0
  g1[4] = (int)(tile_h & 0xFFFFu);              // tile_dim1, tile_dim2=0
  g1[5] = (int)(unsigned)(d0s & 0xFFFFFFFFull); // tensor_dim0_stride lo
  g1[6] = (int)((unsigned)((d0s >> 32) & 0xFFFFull) |
                ((unsigned)(d1s & 0xFFFFull) << 16));  // d0s hi, d1s lo
  g1[7] = (int)(unsigned)((d1s >> 16) & 0xFFFFFFFFull);
  i32x4 gz4;
  gz4[0] = 0; gz4[1] = 0; gz4[2] = 0; gz4[3] = 0;
  i32x8 gz8;
#pragma unroll
  for (int i = 0; i < 8; ++i) gz8[i] = 0;
  __builtin_amdgcn_tensor_load_to_lds(g0, g1, gz4, gz4, gz8, 0);
}
#endif

// ---------------------------------------------------------------------------
// fp32 -> bf16 conversion
// ---------------------------------------------------------------------------
__global__ void cvt_f32_bf16_k(const float* __restrict__ s,
                               __bf16* __restrict__ d, int n) {
  int i = blockIdx.x * blockDim.x + threadIdx.x;
  if (i < n) d[i] = (__bf16)s[i];
}

// ---------------------------------------------------------------------------
// Offset conv: implicit GEMM.  out[m, n] = sum_k ow[m, k] * im2col(x)[k, n]
//   m in [0, 2*K*K), n = b*4096 + y*64 + x, k = c*K*K + ki*K + kj, Kd = 64*K*K
// One wave per 16-pixel tile; carries all M-tiles so the gathered B fragment
// is built once per k-step and reused across every WMMA.
// ---------------------------------------------------------------------------
template <int K, int DIL>
__global__ __launch_bounds__(128) void offset_gemm_k(
    const __bf16* __restrict__ xbf, const __bf16* __restrict__ wbf,
    const float* __restrict__ bias, float* __restrict__ offs) {
  constexpr int KK = K * K;
  constexpr int Coff = 2 * KK;
  constexpr int Mt = (Coff + 15) / 16;
  constexpr int PAD = ((K - 1) / 2) * DIL;
  constexpr int Kd = 64 * KK;

  const int wave = blockIdx.x * (blockDim.x >> 5) + (threadIdx.x >> 5);
  const int lane = threadIdx.x & 31;
  const int g = lane >> 4;
  const int col = lane & 15;

  const int n = wave * 16 + col;
  const int b = n >> 12;
  const int p = n & 4095;
  const int y = p >> 6, x = p & 63;

  v8f acc[Mt];
#pragma unroll
  for (int i = 0; i < Mt; ++i) acc[i] = vzero8();

  for (int kk = 0; kk < Kd; kk += 32) {
    if (kk + 32 < Kd)
      __builtin_prefetch(wbf + (size_t)col * Kd + kk + 32, 0, 0);

    v16bf vb;
#pragma unroll
    for (int h = 0; h < 16; ++h) {
      int k = kk + h + (g << 4);
      int c = k / KK;
      int t = k - c * KK;
      int ki = t / K, kj = t - ki * K;
      int yy = y - PAD + DIL * ki;
      int xx = x - PAD + DIL * kj;
      float v = 0.0f;
      if ((unsigned)yy < 64u && (unsigned)xx < 64u)
        v = (float)xbf[(((b << 6) + c) << 12) + (yy << 6) + xx];
      vb[h] = (__bf16)v;
    }

#pragma unroll
    for (int mt = 0; mt < Mt; ++mt) {
      int row = mt * 16 + col;
      v16bf va = load_a_rowmajor(wbf, row, Kd, kk, g, row < Coff);
      acc[mt] = wmma_bf16(va, vb, acc[mt]);
    }
  }

#pragma unroll
  for (int mt = 0; mt < Mt; ++mt) {
#pragma unroll
    for (int r = 0; r < 8; ++r) {
      int m = mt * 16 + r + (g << 3);
      if (m < Coff)
        offs[(((size_t)b * Coff + m) << 12) + p] = acc[mt][r] + bias[m];
    }
  }
}

// ---------------------------------------------------------------------------
// Deformable conv: out[b,o,y,x] = sum_{c,t} samp(b,c,t,y,x) * dw[o, c*KK+t]
// One block = 64 consecutive pixels of one batch image. The block's offset
// tile (2*KK x 64 fp32) is staged into LDS with the Tensor Data Mover so the
// inner-loop dy/dx reads are DS ops; the bilinear B fragment is built once
// per k-step and reused across all 4 M-tiles (64 output channels).
// ---------------------------------------------------------------------------
DEVINL float bilin_tap(const __bf16* __restrict__ ch, int iy, int ix) {
  bool ok = ((unsigned)iy < 64u) && ((unsigned)ix < 64u);
  int cy = min(max(iy, 0), 63);
  int cx = min(max(ix, 0), 63);
  float v = (float)ch[(cy << 6) + cx];
  return ok ? v : 0.0f;
}

template <int K, int DIL>
__global__ __launch_bounds__(128) void deform_gemm_k(
    const __bf16* __restrict__ xbf, const __bf16* __restrict__ dwbf,
    const float* __restrict__ offs, float* __restrict__ outraw) {
  constexpr int KK = K * K;
  constexpr int Coff = 2 * KK;
  constexpr int PAD = ((K - 1) / 2) * DIL;
  constexpr int Kd = 64 * KK;

  __shared__ __align__(16) float lds_off[Coff * 64];  // [offset_ch][pixel]

  const int lane = threadIdx.x & 31;
  const int g = lane >> 4;
  const int col = lane & 15;
  const int wl = threadIdx.x >> 5;  // wave in block
  const int pl = wl * 16 + col;     // pixel within block tile

  const int n0 = blockIdx.x * 64;
  const int b = n0 >> 12;
  const int p0 = n0 & 4095;
  const int p = p0 + pl;
  const int y = p >> 6, x = p & 63;

#if HAVE_TDM
  if (threadIdx.x < 32) {
    tdm_load_2d(offs + (((size_t)b * Coff) << 12) + p0, lds_addr_of(lds_off),
                /*tile_w=*/64, /*tile_h=*/Coff, /*row_stride=*/4096,
                /*data_size_log2=*/2);
    __builtin_amdgcn_s_wait_tensorcnt(0);
  }
  __syncthreads();
#else
  for (int i = threadIdx.x; i < Coff * 64; i += 128) {
    int r = i >> 6, c = i & 63;
    lds_off[i] = offs[(((size_t)b * Coff + r) << 12) + p0 + c];
  }
  __syncthreads();
#endif

  v8f acc[4];
#pragma unroll
  for (int i = 0; i < 4; ++i) acc[i] = vzero8();

  for (int kk = 0; kk < Kd; kk += 32) {
    if (kk + 32 < Kd)
      __builtin_prefetch(dwbf + (size_t)col * Kd + kk + 32, 0, 0);

    v16bf vb;
#pragma unroll
    for (int h = 0; h < 16; ++h) {
      int k = kk + h + (g << 4);
      int c = k / KK;
      int t = k - c * KK;
      float dy = lds_off[(2 * t) * 64 + pl];
      float dx = lds_off[(2 * t + 1) * 64 + pl];
      float py = (float)(y - PAD + DIL * (t / K)) + dy;
      float px = (float)(x - PAD + DIL * (t - (t / K) * K)) + dx;
      float y0f = floorf(py), x0f = floorf(px);
      int y0 = (int)y0f, x0 = (int)x0f;
      float wy1 = py - y0f, wx1 = px - x0f;
      float wy0 = 1.0f - wy1, wx0 = 1.0f - wx1;
      const __bf16* ch = xbf + (((size_t)(b << 6) + c) << 12);
      float v = bilin_tap(ch, y0, x0) * (wy0 * wx0) +
                bilin_tap(ch, y0, x0 + 1) * (wy0 * wx1) +
                bilin_tap(ch, y0 + 1, x0) * (wy1 * wx0) +
                bilin_tap(ch, y0 + 1, x0 + 1) * (wy1 * wx1);
      vb[h] = (__bf16)v;
    }

#pragma unroll
    for (int mt = 0; mt < 4; ++mt) {
      v16bf va = load_a_rowmajor(dwbf, mt * 16 + col, Kd, kk, g, true);
      acc[mt] = wmma_bf16(va, vb, acc[mt]);
    }
  }

#pragma unroll
  for (int mt = 0; mt < 4; ++mt) {
#pragma unroll
    for (int r = 0; r < 8; ++r) {
      int m = mt * 16 + r + (g << 3);
      outraw[(((size_t)(b << 6) + m) << 12) + p] = acc[mt][r];
    }
  }
}

// ---------------------------------------------------------------------------
// Fused 1x1 conv: out[m, n] = sum_{k<192} fw[m, k] * cat[b, k, p]
// One block = 64 pixels. Both operands are TDM-staged into LDS:
//   lds_w: fw (64 x 192 bf16, 24KB), lds_x: cat tile (192 x 64 bf16, 24KB).
// Inner loop is pure LDS -> WMMA.
// ---------------------------------------------------------------------------
__global__ __launch_bounds__(128) void fused_gemm_k(
    const __bf16* __restrict__ catbf, const __bf16* __restrict__ fwbf,
    float* __restrict__ outraw) {
  __shared__ __align__(16) __bf16 lds_w[64 * 192];   // [m][k]
  __shared__ __align__(16) __bf16 lds_x[192 * 64];   // [k][pixel]

  const int lane = threadIdx.x & 31;
  const int g = lane >> 4;
  const int col = lane & 15;
  const int wl = threadIdx.x >> 5;
  const int pl = wl * 16 + col;

  const int n0 = blockIdx.x * 64;
  const int b = n0 >> 12;
  const int p0 = n0 & 4095;
  const int p = p0 + pl;

#if HAVE_TDM
  if (threadIdx.x < 32) {
    tdm_load_2d(fwbf, lds_addr_of(lds_w),
                /*tile_w=*/192, /*tile_h=*/64, /*row_stride=*/192,
                /*data_size_log2=*/1);
    tdm_load_2d(catbf + (((size_t)b * 192) << 12) + p0, lds_addr_of(lds_x),
                /*tile_w=*/64, /*tile_h=*/192, /*row_stride=*/4096,
                /*data_size_log2=*/1);
    __builtin_amdgcn_s_wait_tensorcnt(0);
  }
  __syncthreads();
#else
  for (int i = threadIdx.x; i < 64 * 192; i += 128) lds_w[i] = fwbf[i];
  for (int i = threadIdx.x; i < 192 * 64; i += 128) {
    int r = i >> 6, c = i & 63;
    lds_x[i] = catbf[(((size_t)b * 192 + r) << 12) + p0 + c];
  }
  __syncthreads();
#endif

  v8f acc[4];
#pragma unroll
  for (int i = 0; i < 4; ++i) acc[i] = vzero8();

#pragma unroll
  for (int kk = 0; kk < 192; kk += 32) {
    v16bf vb;
#pragma unroll
    for (int h = 0; h < 16; ++h) {
      int k = kk + h + (g << 4);
      vb[h] = lds_x[k * 64 + pl];
    }
#pragma unroll
    for (int mt = 0; mt < 4; ++mt) {
      v16bf va = load_a_rowmajor(lds_w, mt * 16 + col, 192, kk, g, true);
      acc[mt] = wmma_bf16(va, vb, acc[mt]);
    }
  }

#pragma unroll
  for (int mt = 0; mt < 4; ++mt) {
#pragma unroll
    for (int r = 0; r < 8; ++r) {
      int m = mt * 16 + r + (g << 3);
      outraw[(((size_t)(b << 6) + m) << 12) + p] = acc[mt][r];
    }
  }
}

// ---------------------------------------------------------------------------
// BatchNorm statistics over (B, H, W) per channel: mean + rsqrt(var + eps)
// ---------------------------------------------------------------------------
__global__ void bn_stats_k(const float* __restrict__ raw,
                           float* __restrict__ stats) {
  __shared__ float s1[256];
  __shared__ float s2[256];
  const int ch = blockIdx.x;
  float a = 0.0f, q = 0.0f;
  for (int i = threadIdx.x; i < 16384; i += 256) {
    int b = i >> 12, p = i & 4095;
    float v = raw[(((size_t)(b << 6) + ch) << 12) + p];
    a += v;
    q += v * v;
  }
  s1[threadIdx.x] = a;
  s2[threadIdx.x] = q;
  __syncthreads();
  for (int s = 128; s > 0; s >>= 1) {
    if (threadIdx.x < (unsigned)s) {
      s1[threadIdx.x] += s1[threadIdx.x + s];
      s2[threadIdx.x] += s2[threadIdx.x + s];
    }
    __syncthreads();
  }
  if (threadIdx.x == 0) {
    float m = s1[0] * (1.0f / 16384.0f);
    float var = s2[0] * (1.0f / 16384.0f) - m * m;
    stats[ch] = m;
    stats[64 + ch] = rsqrtf(var + 1e-5f);
  }
}

__global__ void bn_relu_tobf_k(const float* __restrict__ raw,
                               const float* __restrict__ stats,
                               const float* __restrict__ gamma,
                               const float* __restrict__ beta,
                               __bf16* __restrict__ catbf, int branch) {
  int i = blockIdx.x * blockDim.x + threadIdx.x;
  int p = i & 4095;
  int ch = (i >> 12) & 63;
  int b = i >> 18;
  float o = gamma[ch] * (raw[i] - stats[ch]) * stats[64 + ch] + beta[ch];
  o = fmaxf(o, 0.0f);
  catbf[(((size_t)b * 192 + branch * 64 + ch) << 12) + p] = (__bf16)o;
}

__global__ void bn_relu_final_k(const float* __restrict__ raw,
                                const float* __restrict__ stats,
                                const float* __restrict__ gamma,
                                const float* __restrict__ beta,
                                float* __restrict__ out) {
  int i = blockIdx.x * blockDim.x + threadIdx.x;
  int ch = (i >> 12) & 63;
  float o = gamma[ch] * (raw[i] - stats[ch]) * stats[64 + ch] + beta[ch];
  out[i] = fmaxf(o, 0.0f);
}

// ---------------------------------------------------------------------------
// Host orchestration
// ---------------------------------------------------------------------------
extern "C" void kernel_launch(void* const* d_in, const int* in_sizes, int n_in,
                              void* d_out, int out_size, void* d_ws,
                              size_t ws_size, hipStream_t stream) {
  (void)in_sizes; (void)n_in; (void)out_size; (void)ws_size;

  const float* x = (const float*)d_in[0];
  const float* ow[3] = {(const float*)d_in[1], (const float*)d_in[6],
                        (const float*)d_in[11]};
  const float* ob[3] = {(const float*)d_in[2], (const float*)d_in[7],
                        (const float*)d_in[12]};
  const float* dw[3] = {(const float*)d_in[3], (const float*)d_in[8],
                        (const float*)d_in[13]};
  const float* gg[3] = {(const float*)d_in[4], (const float*)d_in[9],
                        (const float*)d_in[14]};
  const float* bb[3] = {(const float*)d_in[5], (const float*)d_in[10],
                        (const float*)d_in[15]};
  const float* fw = (const float*)d_in[16];
  const float* fg = (const float*)d_in[17];
  const float* fb = (const float*)d_in[18];

  char* wp = (char*)d_ws;
  auto alloc = [&](size_t bytes) -> char* {
    char* r = wp;
    wp += (bytes + 255) & ~(size_t)255;
    return r;
  };

  const int KKv[3] = {9, 25, 49};
  __bf16* xbf = (__bf16*)alloc((size_t)1048576 * 2);
  __bf16* owbf[3];
  __bf16* dwbf[3];
  for (int i = 0; i < 3; ++i) {
    owbf[i] = (__bf16*)alloc((size_t)128 * KKv[i] * KKv[i] * 2);
    dwbf[i] = (__bf16*)alloc((size_t)4096 * KKv[i] * 2);
  }
  __bf16* fwbf = (__bf16*)alloc((size_t)12288 * 2);
  float* offs = (float*)alloc((size_t)1605632 * 4);
  float* braw = (float*)alloc((size_t)1048576 * 4);
  __bf16* catbf = (__bf16*)alloc((size_t)3145728 * 2);
  float* fraw = (float*)alloc((size_t)1048576 * 4);
  float* stats = (float*)alloc((size_t)128 * 4);

  auto cvt = [&](const float* s, __bf16* d, int n) {
    cvt_f32_bf16_k<<<(n + 255) / 256, 256, 0, stream>>>(s, d, n);
  };
  cvt(x, xbf, 1048576);
  const int owN[3] = {128 * 81, 128 * 625, 128 * 2401};
  const int dwN[3] = {4096 * 9, 4096 * 25, 4096 * 49};
  for (int i = 0; i < 3; ++i) {
    cvt(ow[i], owbf[i], owN[i]);
    cvt(dw[i], dwbf[i], dwN[i]);
  }
  cvt(fw, fwbf, 12288);

  dim3 ggrid(256), gblk(128);

  // Branch 0: K=3, dil=1
  offset_gemm_k<3, 1><<<ggrid, gblk, 0, stream>>>(xbf, owbf[0], ob[0], offs);
  deform_gemm_k<3, 1><<<ggrid, gblk, 0, stream>>>(xbf, dwbf[0], offs, braw);
  bn_stats_k<<<64, 256, 0, stream>>>(braw, stats);
  bn_relu_tobf_k<<<4096, 256, 0, stream>>>(braw, stats, gg[0], bb[0], catbf, 0);

  // Branch 1: K=5, dil=2
  offset_gemm_k<5, 2><<<ggrid, gblk, 0, stream>>>(xbf, owbf[1], ob[1], offs);
  deform_gemm_k<5, 2><<<ggrid, gblk, 0, stream>>>(xbf, dwbf[1], offs, braw);
  bn_stats_k<<<64, 256, 0, stream>>>(braw, stats);
  bn_relu_tobf_k<<<4096, 256, 0, stream>>>(braw, stats, gg[1], bb[1], catbf, 1);

  // Branch 2: K=7, dil=3
  offset_gemm_k<7, 3><<<ggrid, gblk, 0, stream>>>(xbf, owbf[2], ob[2], offs);
  deform_gemm_k<7, 3><<<ggrid, gblk, 0, stream>>>(xbf, dwbf[2], offs, braw);
  bn_stats_k<<<64, 256, 0, stream>>>(braw, stats);
  bn_relu_tobf_k<<<4096, 256, 0, stream>>>(braw, stats, gg[2], bb[2], catbf, 2);

  // Fused 1x1 conv + final BN + ReLU
  fused_gemm_k<<<ggrid, gblk, 0, stream>>>(catbf, fwbf, fraw);
  bn_stats_k<<<64, 256, 0, stream>>>(fraw, stats);
  bn_relu_final_k<<<4096, 256, 0, stream>>>(fraw, stats, fg, fb,
                                            (float*)d_out);
}
